// SwinAttention_5995774345646
// MI455X (gfx1250) — compile-verified, matrix operand order checked by reading, not run on hardware
//
#include <hip/hip_runtime.h>
#include <cstdint>

// ---------------------------------------------------------------------------
// Swin shifted-window attention block for gfx1250 (MI455X), bf16 WMMA path.
// B=32, C=128, H=W=112, heads=4, d=32, M=7 (N=49 padded to 64 rows/window).
// Workspace layout (~357 MB):
//   [0]                xw   : bf16, 8192 win x 64 rows x 128    (128 MB)
//   [+134217728]       Z    : f32,  8192 win x 49 rows x 128    (196 MB)
//   [+339738624]       comb : f32,  256 wnd x 4 h x 64 x 64     ( 16 MB)
//   [+356515840]       qwb  : bf16, 384 x 128
//   [+356614144]       pwb  : bf16, 128 x 128
// ---------------------------------------------------------------------------

typedef __bf16 bf16;
typedef __attribute__((ext_vector_type(16))) __bf16   v16bf;
typedef __attribute__((ext_vector_type(8)))  float    v8f;
typedef __attribute__((ext_vector_type(4)))  uint32_t v4u;
typedef __attribute__((ext_vector_type(8)))  uint32_t v8u;

__device__ __forceinline__ bf16 f2bf(float f) {
  union { float f; uint32_t u; } x; x.f = f;
  uint32_t r = x.u + 0x7FFFu + ((x.u >> 16) & 1u);   // round-to-nearest-even
  uint16_t h = (uint16_t)(r >> 16);
  return __builtin_bit_cast(bf16, h);
}

__device__ __forceinline__ v8f vzero8() {
  v8f z;
#pragma unroll
  for (int i = 0; i < 8; ++i) z[i] = 0.0f;
  return z;
}

// WMMA bf16 A-fragment (16x32 MxK) / B-fragment (32x16 KxN) per CDNA5 ISA
// striping: lane L holds row (A) / column (B) = L&15; lane group g = L>>4
// holds two contiguous 8-element K chunks at offsets g*8 and 16+g*8.
__device__ __forceinline__ v16bf load_frag(const bf16* __restrict__ row,
                                           int kb, int grp) {
  const bf16* p0 = row + kb + grp * 8;
  const bf16* p1 = p0 + 16;
  v16bf a;
#pragma unroll
  for (int i = 0; i < 8; ++i) { a[i] = p0[i]; a[8 + i] = p1[i]; }
  return a;
}

__device__ __forceinline__ v8f wmma_bf16(v16bf a, v16bf b, v8f c) {
  return __builtin_amdgcn_wmma_f32_16x16x32_bf16(
      /*neg_a=*/false, a, /*neg_b=*/false, b,
      /*c_mod=*/(short)0, c, /*reuse_a=*/false, /*reuse_b=*/false);
}

// ---------------------------------------------------------------------------
// Kernel: convert qkv_w (384x128) and proj_w (128x128) to bf16.
__global__ __launch_bounds__(256) void k_prep_w(
    const float* __restrict__ qw, const float* __restrict__ pwf,
    bf16* __restrict__ qwb, bf16* __restrict__ pwb) {
  int i = blockIdx.x * 256 + threadIdx.x;
  if (i < 384 * 128) qwb[i] = f2bf(qw[i]);
  if (i < 128 * 128) pwb[i] = f2bf(pwf[i]);
}

// Kernel: comb[wnd][h][n(64)][m(64)] = rel-pos-bias + shift-mask, -1e30 pads.
__global__ __launch_bounds__(256) void k_prep_comb(
    const float* __restrict__ bias_table, const int* __restrict__ rel_index,
    const float* __restrict__ attn_mask, float* __restrict__ comb) {
  int wnd = blockIdx.x;                       // 256 windows per image
  for (int t = threadIdx.x; t < 4 * 64 * 64; t += 256) {
    int h = t >> 12, rem = t & 4095, n = rem >> 6, m = rem & 63;
    float v;
    if (n >= 49 || m >= 49) v = -1.0e30f;
    else v = bias_table[rel_index[n * 49 + m] * 4 + h] +
             attn_mask[(size_t)wnd * 2401 + n * 49 + m];
    comb[((size_t)wnd * 4 + h) * 4096 + (n << 6) + m] = v;
  }
}

// Kernel: zero the 15 pad rows (49..63) of each window.
__global__ __launch_bounds__(128) void k_zero_pad(bf16* __restrict__ xw) {
  size_t base = ((size_t)blockIdx.x * 64 + 49) * 128;
  bf16 z = f2bf(0.0f);
  for (int i = threadIdx.x; i < 15 * 128; i += 128) xw[base + i] = z;
}

// ---------------------------------------------------------------------------
// Kernel 1: LayerNorm over C + cyclic shift (-3,-3) + window partition -> bf16
// One block handles 16 W-positions x 128 channels (LDS transpose for LN).
__global__ __launch_bounds__(128) void k1_ln(
    const float* __restrict__ x, const float* __restrict__ nw,
    const float* __restrict__ nb, bf16* __restrict__ xw) {
  __shared__ float tile[128][17];
  __shared__ float smu[16], srs[16];
  int blk = blockIdx.x;
  int wt = blk % 7;            // W tile (16 positions)
  int h  = (blk / 7) % 112;
  int b  = blk / (7 * 112);
  int w0 = wt * 16;
  int c  = threadIdx.x;        // channel

  const float* px = x + (((size_t)b * 128 + c) * 112 + h) * 112 + w0;
#pragma unroll
  for (int i = 0; i < 16; ++i) tile[c][i] = px[i];
  __syncthreads();

  if (c < 16) {
    float s = 0.f, s2 = 0.f;
    for (int cc = 0; cc < 128; ++cc) {
      float v = tile[cc][c]; s += v; s2 += v * v;
    }
    float mu  = s * (1.0f / 128.0f);
    float var = s2 * (1.0f / 128.0f) - mu * mu;
    smu[c] = mu;
    srs[c] = rsqrtf(var + 1e-5f);
  }
  __syncthreads();

  float g = nw[c], be = nb[c];
#pragma unroll
  for (int i = 0; i < 16; ++i) {
    int w = w0 + i;
    float v = (tile[c][i] - smu[i]) * srs[i] * g + be;
    int h2 = (h + 109) % 112;            // shift by -3
    int w2 = (w + 109) % 112;
    int wh = h2 / 7, ww = w2 / 7;
    int r  = (h2 % 7) * 7 + (w2 % 7);
    size_t win = ((size_t)b * 16 + wh) * 16 + ww;
    xw[(win * 64 + r) * 128 + c] = f2bf(v);
  }
}

// ---------------------------------------------------------------------------
// Kernel 2: one workgroup (256 thr = 8 waves) per window.
//   TDM X-tile load -> QKV GEMM -> QK^T + softmax -> P*V -> proj GEMM.
__global__ __launch_bounds__(256) void k2_attn(
    const bf16*  __restrict__ xw,   const bf16*  __restrict__ qwb,
    const bf16*  __restrict__ pwb,  const float* __restrict__ qkvb,
    const float* __restrict__ pb,   const float* __restrict__ comb,
    float* __restrict__ Z) {
  __shared__ bf16 sX[64][128];        // X tile; reused for O (n, h*32+d)
  __shared__ bf16 sQ[4][64][32];      // (h, n, d)  d-contiguous
  __shared__ bf16 sK[4][64][32];      // (h, m, d)  d-contiguous
  __shared__ bf16 sVT[4][32][64];     // (h, d, m)  m-contiguous
  __shared__ bf16 sP[4][64][64];      // (h, n, m)  m-contiguous

  const int win  = blockIdx.x;
  const int wnd  = win & 255;
  const int tid  = threadIdx.x;
  const int wv   = tid >> 5;
  const int lane = tid & 31;
  const int l16  = lane & 15;
  const int grp  = lane >> 4;

  const float* cbase = comb + (((size_t)wnd * 4 + (wv >> 1)) * 4096);
  __builtin_prefetch(cbase, 0, 3);    // global_prefetch of our mask tile

  // ---- Phase A: TDM async load of the 64x128 bf16 X tile into LDS --------
  if (wv == 0) {
    uint64_t ga = (uint64_t)(const void*)(xw + (size_t)win * 64 * 128);
    uint32_t lb = (uint32_t)(uint64_t)(const void*)&sX[0][0];
    v4u g0; v8u g1; v4u g2; v4u g3;
    g0[0] = 1u;                                   // count=1, user D#
    g0[1] = lb;                                   // lds_addr
    g0[2] = (uint32_t)ga;                         // global_addr[31:0]
    g0[3] = ((uint32_t)(ga >> 32) & 0x01FFFFFFu)  // global_addr[56:32]
            | (2u << 30);                         // type=2 (image)
    g1[0] = (1u << 16);          // data_size=1 (2 bytes), no multicast
    g1[1] = (128u << 16);        // tensor_dim0 = 128 elements
    g1[2] = (64u << 16);         // tensor_dim1 = 64 rows
    g1[3] = (128u << 16);        // tile_dim0   = 128
    g1[4] = 64u;                 // tile_dim1   = 64 (tile_dim2 = 0)
    g1[5] = 128u;                // tensor_dim0_stride = 128
    g1[6] = (8192u << 16);       // tensor_dim1_stride = 64*128
    g1[7] = 0u;
    g2[0] = g2[1] = g2[2] = g2[3] = 0u;
    g3[0] = g3[1] = g3[2] = g3[3] = 0u;
    asm volatile("tensor_load_to_lds %0, %1, %2, %3"
                 :: "s"(g0), "s"(g1), "s"(g2), "s"(g3) : "memory");
    __builtin_amdgcn_s_wait_tensorcnt(0);
  }
  __syncthreads();

  // ---- Phase B: QKV GEMM  (64x384) = sX(64x128) * qkv_w^T ----------------
  {
    const int jb = wv * 48;                 // 3 col tiles per wave
    v8f acc[4][3];
#pragma unroll
    for (int mt = 0; mt < 4; ++mt)
#pragma unroll
      for (int ct = 0; ct < 3; ++ct) acc[mt][ct] = vzero8();

    for (int kb = 0; kb < 128; kb += 32) {
      v16bf af[4];
#pragma unroll
      for (int mt = 0; mt < 4; ++mt)
        af[mt] = load_frag(&sX[mt * 16 + l16][0], kb, grp);
#pragma unroll
      for (int ct = 0; ct < 3; ++ct) {
        int j = jb + ct * 16 + l16;
        v16bf bfr = load_frag(qwb + (size_t)j * 128, kb, grp);
#pragma unroll
        for (int mt = 0; mt < 4; ++mt)
          acc[mt][ct] = wmma_bf16(af[mt], bfr, acc[mt][ct]);
      }
    }
    const float qscale = 0.17677669529663687f;   // 1/sqrt(32)
#pragma unroll
    for (int mt = 0; mt < 4; ++mt) {
#pragma unroll
      for (int ct = 0; ct < 3; ++ct) {
        int j = jb + ct * 16 + l16;
        float bia = qkvb[j];
#pragma unroll
        for (int r = 0; r < 8; ++r) {
          int nrow = mt * 16 + r + 8 * grp;
          float v = acc[mt][ct][r] + bia;
          if (j < 128) {
            sQ[j >> 5][nrow][j & 31] = f2bf(v * qscale);
          } else if (j < 256) {
            int jj = j - 128;
            sK[jj >> 5][nrow][jj & 31] = f2bf(v);
          } else {
            int jj = j - 256;
            sVT[jj >> 5][jj & 31][nrow] = f2bf(v);    // V transposed
          }
        }
      }
    }
  }
  __syncthreads();

  // ---- Phase C: S = Q*K^T (+bias+mask) -> softmax -> sP (bf16) -----------
  const int hh = wv >> 1;        // head
  const int hf = wv & 1;         // half of the 64 rows
  {
    v8f sc[2][4];
#pragma unroll
    for (int i = 0; i < 2; ++i) {
      int mt = hf * 2 + i;
      v16bf aq = load_frag(&sQ[hh][mt * 16 + l16][0], 0, grp);
#pragma unroll
      for (int ct = 0; ct < 4; ++ct) {
        v16bf bk = load_frag(&sK[hh][ct * 16 + l16][0], 0, grp);
        sc[i][ct] = wmma_bf16(aq, bk, vzero8());     // K = d = 32: one WMMA
      }
    }
#pragma unroll
    for (int i = 0; i < 2; ++i) {
      int mt = hf * 2 + i;
#pragma unroll
      for (int r = 0; r < 8; ++r) {
        int nrow = mt * 16 + r + 8 * grp;
        float rowmax = -3.0e38f;
#pragma unroll
        for (int ct = 0; ct < 4; ++ct) {
          int mcol = ct * 16 + l16;
          float s = sc[i][ct][r] + cbase[nrow * 64 + mcol];
          sc[i][ct][r] = s;
          rowmax = fmaxf(rowmax, s);
        }
#pragma unroll
        for (int off = 1; off < 16; off <<= 1)
          rowmax = fmaxf(rowmax, __shfl_xor(rowmax, off, 32));
        float ev[4], rs = 0.f;
#pragma unroll
        for (int ct = 0; ct < 4; ++ct) {
          float e = __expf(sc[i][ct][r] - rowmax);
          ev[ct] = e; rs += e;
        }
#pragma unroll
        for (int off = 1; off < 16; off <<= 1)
          rs += __shfl_xor(rs, off, 32);
        float rinv = 1.0f / rs;
#pragma unroll
        for (int ct = 0; ct < 4; ++ct) {
          int mcol = ct * 16 + l16;
          sP[hh][nrow][mcol] = f2bf(ev[ct] * rinv);
        }
      }
    }
  }
  __syncthreads();

  // ---- Phase D: O = P * V  (64x32 per head), store into sX as (n, h*32+d) -
  {
    v8f oacc[2][2];
#pragma unroll
    for (int i = 0; i < 2; ++i)
#pragma unroll
      for (int nt = 0; nt < 2; ++nt) oacc[i][nt] = vzero8();
    for (int kb = 0; kb < 64; kb += 32) {
      v16bf ap[2];
#pragma unroll
      for (int i = 0; i < 2; ++i)
        ap[i] = load_frag(&sP[hh][(hf * 2 + i) * 16 + l16][0], kb, grp);
#pragma unroll
      for (int nt = 0; nt < 2; ++nt) {
        v16bf bv = load_frag(&sVT[hh][nt * 16 + l16][0], kb, grp);
#pragma unroll
        for (int i = 0; i < 2; ++i)
          oacc[i][nt] = wmma_bf16(ap[i], bv, oacc[i][nt]);
      }
    }
#pragma unroll
    for (int i = 0; i < 2; ++i)
#pragma unroll
      for (int nt = 0; nt < 2; ++nt)
#pragma unroll
        for (int r = 0; r < 8; ++r) {
          int nrow = (hf * 2 + i) * 16 + r + 8 * grp;
          int dcol = nt * 16 + l16;
          sX[nrow][hh * 32 + dcol] = f2bf(oacc[i][nt][r]);
        }
  }
  __syncthreads();

  // ---- Phase E: proj GEMM  Z(64x128) = O(64x128) * proj_w^T + proj_b -----
  {
    const int mt = wv >> 1;
    const int j0 = (wv & 1) * 64;
    v8f zacc[4];
#pragma unroll
    for (int ci = 0; ci < 4; ++ci) zacc[ci] = vzero8();
    for (int kb = 0; kb < 128; kb += 32) {
      v16bf ao = load_frag(&sX[mt * 16 + l16][0], kb, grp);
#pragma unroll
      for (int ci = 0; ci < 4; ++ci) {
        int jc = j0 + ci * 16 + l16;
        v16bf bw = load_frag(pwb + (size_t)jc * 128, kb, grp);
        zacc[ci] = wmma_bf16(ao, bw, zacc[ci]);
      }
    }
#pragma unroll
    for (int ci = 0; ci < 4; ++ci) {
      int jc = j0 + ci * 16 + l16;
      float bb = pb[jc];
#pragma unroll
      for (int r = 0; r < 8; ++r) {
        int nrow = mt * 16 + r + 8 * grp;
        if (nrow < 49)
          Z[((size_t)win * 49 + nrow) * 128 + jc] = zacc[ci][r] + bb;
      }
    }
  }
}

// ---------------------------------------------------------------------------
// Kernel 3: window reverse + un-shift (+3,+3) + residual, out (B,C,H,W).
__global__ __launch_bounds__(256) void k3_out(
    const float* __restrict__ x, const float* __restrict__ Z,
    float* __restrict__ out) {
  size_t idx = (size_t)blockIdx.x * 256 + threadIdx.x;
  if (idx >= (size_t)32 * 128 * 112 * 112) return;
  int w = (int)(idx % 112); size_t t = idx / 112;
  int h = (int)(t % 112);   t /= 112;
  int c = (int)(t % 128);
  int b = (int)(t / 128);
  int h2 = (h + 109) % 112;      // inverse of +3 roll
  int w2 = (w + 109) % 112;
  int wh = h2 / 7, ww = w2 / 7;
  int r  = (h2 % 7) * 7 + (w2 % 7);
  size_t win = ((size_t)b * 16 + wh) * 16 + ww;
  out[idx] = x[idx] + Z[(win * 49 + r) * 128 + c];
}

// ---------------------------------------------------------------------------
extern "C" void kernel_launch(void* const* d_in, const int* in_sizes, int n_in,
                              void* d_out, int out_size, void* d_ws,
                              size_t ws_size, hipStream_t stream) {
  (void)in_sizes; (void)n_in; (void)out_size; (void)ws_size;
  const float* x     = (const float*)d_in[0];
  const float* nw    = (const float*)d_in[1];
  const float* nb    = (const float*)d_in[2];
  const float* qkvw  = (const float*)d_in[3];
  const float* qkvb  = (const float*)d_in[4];
  const float* projw = (const float*)d_in[5];
  const float* projb = (const float*)d_in[6];
  const float* btab  = (const float*)d_in[7];
  const int*   ridx  = (const int*)d_in[8];
  const float* amask = (const float*)d_in[9];
  float* out = (float*)d_out;

  char* wsb = (char*)d_ws;
  bf16*  xw   = (bf16*)wsb;                               // 134,217,728 B
  float* Zb   = (float*)(wsb + 134217728u);               // 205,520,896 B
  float* comb = (float*)(wsb + 134217728u + 205520896u);  //  16,777,216 B
  bf16*  qwb  = (bf16*)(wsb + 134217728u + 205520896u + 16777216u);
  bf16*  pwb  = qwb + 384 * 128;

  k_prep_w   <<<192,   256, 0, stream>>>(qkvw, projw, qwb, pwb);
  k_prep_comb<<<256,   256, 0, stream>>>(btab, ridx, amask, comb);
  k_zero_pad <<<8192,  128, 0, stream>>>(xw);
  k1_ln      <<<25088, 128, 0, stream>>>(x, nw, nb, xw);
  k2_attn    <<<8192,  256, 0, stream>>>(xw, qwb, pwb, qkvb, projb, comb, Zb);
  k3_out     <<<200704,256, 0, stream>>>(x, Zb, out);
}
